// GeometricAttention_18339510354278
// MI455X (gfx1250) — matrix operand order repaired and checked by reading, not compiled, not run on hardware
//
#include <hip/hip_runtime.h>

typedef __attribute__((ext_vector_type(16))) _Float16 v16h;
typedef __attribute__((ext_vector_type(8)))  _Float16 v8h;
typedef __attribute__((ext_vector_type(8)))  float    v8f;

__device__ __forceinline__ v16h cat8(v8h lo, v8h hi) {
    return __builtin_shufflevector(lo, hi, 0,1,2,3,4,5,6,7,8,9,10,11,12,13,14,15);
}

// CDNA5 async global->LDS copy (16B per lane, tracked with ASYNCcnt).
// vdst = LDS byte offset (low 32 bits of a shared pointer), vaddr = 64-bit global addr.
__device__ __forceinline__ void async_copy_b128(void* lds_dst, const void* gsrc) {
    uint32_t loff = (uint32_t)(uintptr_t)lds_dst;
    asm volatile("global_load_async_to_lds_b128 %0, %1, off"
                 :: "v"(loff), "v"(gsrc) : "memory");
}
__device__ __forceinline__ void wait_async0() {
    asm volatile("s_wait_asynccnt 0x0" ::: "memory");
}

// ---------------- prep kernels ----------------
// pf  : positions f16, row-major [N][512]
// vfT : values f16, per-32-row tile transposed: vfT[tile][d][nl], tile=n>>5, nl=n&31
__global__ void prep_convert(const float* __restrict__ pos, const float* __restrict__ val,
                             _Float16* __restrict__ pf, _Float16* __restrict__ vfT, int n) {
    int i = blockIdx.x * blockDim.x + threadIdx.x;
    if (i < n) {
        pf[i] = (_Float16)pos[i];
        int nrow = i >> 9;
        int d    = i & 511;
        vfT[((size_t)(nrow >> 5) << 14) + (d << 5) + (nrow & 31)] = (_Float16)val[i];
    }
}

__global__ void prep_rowstats(const float* __restrict__ pos, const float* __restrict__ temp,
                              float* __restrict__ psq, float* __restrict__ itau) {
    __shared__ float red[256];
    const int n = blockIdx.x;
    const float* row = pos + (size_t)n * 512;
    float s = 0.f;
    for (int k = threadIdx.x; k < 512; k += 256) { float v = row[k]; s += v * v; }
    red[threadIdx.x] = s;
    __syncthreads();
    for (int off = 128; off > 0; off >>= 1) {
        if (threadIdx.x < off) red[threadIdx.x] += red[threadIdx.x + off];
        __syncthreads();
    }
    if (threadIdx.x == 0) {
        psq[n]  = red[0];
        itau[n] = 1.0f / (fabsf(temp[n]) + 0.1f);
    }
}

// ---------------- main fused kernel ----------------
#define D_DIM   512
#define BM      64
#define BN      32
#define NTILES  128
#define TILE_H  (BN * D_DIM)   // 16384 halfs per staged tile

__launch_bounds__(256)
__global__ void geo_attn(const float*    __restrict__ x,
                         const _Float16* __restrict__ pf,
                         const _Float16* __restrict__ vfT,
                         const float*    __restrict__ psq,
                         const float*    __restrict__ itau,
                         float*          __restrict__ out) {
    extern __shared__ char smem[];
    _Float16* xs   = (_Float16*)smem;            // [64][512]   x block (f16)
    _Float16* psb0 = xs + 32768;                 // [32][512]   P tile buf 0
    _Float16* psb1 = psb0 + TILE_H;              //             P tile buf 1
    _Float16* vsb0 = psb1 + TILE_H;              // [512][32]   V tile buf 0 (pre-transposed)
    _Float16* vsb1 = vsb0 + TILE_H;              //             V tile buf 1
    _Float16* wq   = vsb1 + TILE_H;              // [64][32]    probabilities
    float*    xsq  = (float*)(wq + 64 * 32);     // [64]
    float*    maxb = xsq + 64;                   // [2][64]
    float*    lb   = maxb + 128;                 // [2][64]

    const int tid  = threadIdx.x;
    const int lane = tid & 31;
    const int wave = tid >> 5;
    const int mw   = wave >> 1;      // 0..3  (M stripe)
    const int dw   = wave & 1;       // 0..1  (D half)
    const int h    = lane >> 4;
    const int l15  = lane & 15;
    const int mloc = mw * 16;

    const size_t rowBase = (size_t)blockIdx.x * BM;
    const float* xblk = x + rowBase * D_DIM;

    // ---- prologue: issue async stage of tile 0, then stage x block ----
    {
        const v8h* sp = (const v8h*)pf;          // tile 0
        const v8h* sv = (const v8h*)vfT;
        #pragma unroll
        for (int j = 0; j < 8; ++j) {
            int i = tid + j * 256;
            async_copy_b128(psb0 + (size_t)i * 8, sp + i);
            async_copy_b128(vsb0 + (size_t)i * 8, sv + i);
        }
    }
    if (tid < 64) xsq[tid] = 0.f;
    __syncthreads();
    for (int i = tid; i < (BM * D_DIM) / 4; i += 256) {   // 8192 float4s
        float4 v = ((const float4*)xblk)[i];
        _Float16* d = xs + (size_t)i * 4;
        d[0] = (_Float16)v.x; d[1] = (_Float16)v.y;
        d[2] = (_Float16)v.z; d[3] = (_Float16)v.w;
        float s = v.x * v.x + v.y * v.y + v.z * v.z + v.w * v.w;
        atomicAdd(&xsq[(i * 4) >> 9], s);                 // ds_add_f32
    }

    // ---- per-wave running state ----
    float m_run[8], l_run[8];
    v8f Oacc[16];
    v8f zero = {};
    #pragma unroll
    for (int g = 0; g < 8; ++g) { m_run[g] = -3.0e38f; l_run[g] = 0.f; }
    #pragma unroll
    for (int t = 0; t < 16; ++t) Oacc[t] = zero;

    for (int it = 0; it < NTILES; ++it) {
        const int p = it & 1;
        _Float16* ps = p ? psb1 : psb0;
        _Float16* vs = p ? vsb1 : vsb0;

        // tile `it` must be resident in LDS for every wave
        wait_async0();
        __syncthreads();

        // issue async stage of tile it+1 into the other buffer (overlaps compute)
        if (it + 1 < NTILES) {
            _Float16* psn = p ? psb0 : psb1;
            _Float16* vsn = p ? vsb0 : vsb1;
            const v8h* sp = (const v8h*)(pf  + (size_t)(it + 1) * TILE_H);
            const v8h* sv = (const v8h*)(vfT + (size_t)(it + 1) * TILE_H);
            #pragma unroll
            for (int j = 0; j < 8; ++j) {
                int i = tid + j * 256;
                async_copy_b128(psn + (size_t)i * 8, sp + i);
                async_copy_b128(vsn + (size_t)i * 8, sv + i);
            }
        }

        // ---- GEMM1: S(16x16) = X_stripe . P_sub^T, K = 512 ----
        v8f Sacc = zero;
        {
            const _Float16* arow = xs + (size_t)(mloc + l15) * D_DIM;
            const _Float16* brow = ps + (size_t)(dw * 16 + l15) * D_DIM;
            #pragma unroll
            for (int k0 = 0; k0 < D_DIM; k0 += 32) {
                v8h alo = *(const v8h*)(arow + k0 + 8 * h);
                v8h ahi = *(const v8h*)(arow + k0 + 16 + 8 * h);
                v16h a  = cat8(alo, ahi);
                v8h blo = *(const v8h*)(brow + k0 + 8 * h);
                v8h bhi = *(const v8h*)(brow + k0 + 16 + 8 * h);
                v16h b  = cat8(blo, bhi);
                Sacc = __builtin_amdgcn_wmma_f32_16x16x32_f16(
                    false, a, false, b, (short)0, Sacc, false, false);
            }
        }

        // ---- distances -> logits -> online softmax ----
        const int n0 = it * BN;
        const int   ng = n0 + dw * 16 + l15;
        const float pq = psq[ng];
        const float tv = itau[ng];
        float lg[8];
        #pragma unroll
        for (int g = 0; g < 8; ++g) {
            float sq = xsq[mloc + g + 8 * h] + pq - 2.0f * Sacc[g];
            float d  = sqrtf(fmaxf(sq, 0.0f));
            lg[g] = -d * tv;
        }
        float mx[8];
        #pragma unroll
        for (int g = 0; g < 8; ++g) {
            float v = lg[g];
            v = fmaxf(v, __shfl_xor(v, 1));
            v = fmaxf(v, __shfl_xor(v, 2));
            v = fmaxf(v, __shfl_xor(v, 4));
            v = fmaxf(v, __shfl_xor(v, 8));
            mx[g] = v;
        }
        if (l15 == 0) {
            #pragma unroll
            for (int g = 0; g < 8; ++g)
                maxb[dw * 64 + mloc + g + 8 * h] = mx[g];
        }
        __syncthreads();

        float w[8];
        #pragma unroll
        for (int g = 0; g < 8; ++g) {
            const int mrow = mloc + g + 8 * h;
            float tm = fmaxf(maxb[mrow], maxb[64 + mrow]);
            float mn = fmaxf(m_run[g], tm);
            float alpha = __expf(m_run[g] - mn);
            m_run[g] = mn;
            l_run[g] *= alpha;
            w[g] = __expf(lg[g] - mn);
            #pragma unroll
            for (int dt = 0; dt < 16; ++dt) Oacc[dt][g] *= alpha;
        }
        #pragma unroll
        for (int g = 0; g < 8; ++g) {
            float v = w[g];
            v += __shfl_xor(v, 1);
            v += __shfl_xor(v, 2);
            v += __shfl_xor(v, 4);
            v += __shfl_xor(v, 8);
            l_run[g] += v;
        }
        #pragma unroll
        for (int g = 0; g < 8; ++g)
            wq[(size_t)(mloc + g + 8 * h) * BN + dw * 16 + l15] = (_Float16)w[g];
        __syncthreads();

        // ---- GEMM2: O(16 x 256) += W(16x32) . V(32x256), K = 32 ----
        {
            const _Float16* arow = wq + (size_t)(mloc + l15) * BN;
            v8h alo = *(const v8h*)(arow + 8 * h);
            v8h ahi = *(const v8h*)(arow + 16 + 8 * h);
            v16h a  = cat8(alo, ahi);
            #pragma unroll
            for (int dt = 0; dt < 16; ++dt) {
                const _Float16* brow = vs + (size_t)((dw * 16 + dt) * 16 + l15) * BN;
                v8h blo = *(const v8h*)(brow + 8 * h);
                v8h bhi = *(const v8h*)(brow + 16 + 8 * h);
                v16h b  = cat8(blo, bhi);
                Oacc[dt] = __builtin_amdgcn_wmma_f32_16x16x32_f16(
                    false, a, false, b, (short)0, Oacc[dt], false, false);
            }
        }
    }

    // ---- finalize: combine l across dw pair, normalize, store ----
    if (l15 == 0) {
        #pragma unroll
        for (int g = 0; g < 8; ++g)
            lb[dw * 64 + mloc + g + 8 * h] = l_run[g];
    }
    __syncthreads();
    float scale[8];
    #pragma unroll
    for (int g = 0; g < 8; ++g) {
        const int mrow = mloc + g + 8 * h;
        float ltot = lb[mrow] + lb[64 + mrow];
        float em   = __expf(m_run[g]);
        scale[g]   = em / (ltot * em + 1e-8f);   // reference's +1e-8 applies to true sum
    }
    #pragma unroll
    for (int dt = 0; dt < 16; ++dt) {
        const int dcol = (dw * 16 + dt) * 16 + l15;
        #pragma unroll
        for (int g = 0; g < 8; ++g) {
            const size_t r = rowBase + mloc + g + 8 * h;
            out[r * D_DIM + dcol] = Oacc[dt][g] * scale[g];
        }
    }
}

// ---------------- host launcher ----------------
extern "C" void kernel_launch(void* const* d_in, const int* in_sizes, int n_in,
                              void* d_out, int out_size, void* d_ws, size_t ws_size,
                              hipStream_t stream) {
    const float* x    = (const float*)d_in[0];   // [8,2048,512]
    const float* pos  = (const float*)d_in[1];   // [4096,512]
    const float* val  = (const float*)d_in[2];   // [4096,512]
    const float* temp = (const float*)d_in[3];   // [4096]
    float* out = (float*)d_out;

    char* ws = (char*)d_ws;
    _Float16* pf   = (_Float16*)(ws);                       // 4 MB
    _Float16* vfT  = (_Float16*)(ws + 4194304);             // 4 MB (tile-transposed)
    float*    psq  = (float*)(ws + 8388608);                // 16 KB
    float*    itau = (float*)(ws + 8388608 + 16384);        // 16 KB

    const int NE = 4096 * 512;  // 2097152
    prep_convert<<<NE / 256, 256, 0, stream>>>(pos, val, pf, vfT, NE);
    prep_rowstats<<<4096, 256, 0, stream>>>(pos, temp, psq, itau);

    // LDS: xs 64K + 2x ps 32K + 2x vs 32K + wq 4K + stats 1.25K = 201984 B (< 320K/WGP)
    const size_t lds = 201984;
    geo_attn<<<16384 / 64, 256, lds, stream>>>(x, pf, vfT, psq, itau, out);
}
